// MultiheadAttention_16947940950480
// MI455X (gfx1250) — compile-verified, matrix operand order checked by reading, not run on hardware
//
#include <hip/hip_runtime.h>

#define BQ   4
#define SEQ  2048
#define FDIM 1024
#define DK   1024

#define AS1 __attribute__((address_space(1)))
#define AS3 __attribute__((address_space(3)))

#if defined(__HIP_DEVICE_COMPILE__) && \
    __has_builtin(__builtin_amdgcn_global_load_async_to_lds_b128)
#define HAS_ASYNC_LDS 1
#else
#define HAS_ASYNC_LDS 0
#endif

typedef __attribute__((ext_vector_type(16))) __bf16       bf16x16;
typedef __attribute__((ext_vector_type(8)))  float        f32x8;
typedef __attribute__((ext_vector_type(8)))  unsigned int u32x8;
typedef __attribute__((__vector_size__(16))) int          v4i;

__device__ __forceinline__ unsigned short f2bf(float f) {
  unsigned int u = __builtin_bit_cast(unsigned int, f);
  u += 0x7fffu + ((u >> 16) & 1u);          // round-to-nearest-even
  return (unsigned short)(u >> 16);
}

__device__ __forceinline__ void wait_async_zero() {
#if defined(__HIP_DEVICE_COMPILE__)
#if __has_builtin(__builtin_amdgcn_s_wait_asynccnt)
  __builtin_amdgcn_s_wait_asynccnt(0);
#else
  asm volatile("s_wait_asynccnt 0x0" ::: "memory");
#endif
#endif
}

// Load one 16x32 bf16 WMMA operand fragment for this lane from an LDS tile
// whose rows are 32 bf16 (64B) wide.  Per the CDNA5 ISA layout, lane L
// (L<16) holds row M=L with K in {0..7, 16..23}; lane L+16 holds row M=L
// with K in {8..15, 24..31} -> two contiguous 16-byte reads per lane.
__device__ __forceinline__ bf16x16 load_frag(const unsigned short* p) {
  const uint4 lo = *(const uint4*)(p);
  const uint4 hi = *(const uint4*)(p + 16);
  u32x8 t;
  t[0] = lo.x; t[1] = lo.y; t[2] = lo.z; t[3] = lo.w;
  t[4] = hi.x; t[5] = hi.y; t[6] = hi.z; t[7] = hi.w;
  return __builtin_bit_cast(bf16x16, t);
}

// C[M,N] = scale * sum_k A[m,k]*Bt[n,k]  (+ bias[n])
// A: row-major [M,Kdim] bf16, Bt: row-major [N,Kdim] bf16.
// Block tile 128x128, 8 waves, each wave 32x64 (2x4 WMMA 16x16 tiles).
// Double-buffered LDS; global->LDS staging uses CDNA5 async-to-LDS when
// available (ASYNCcnt), overlapping the next tile's loads with WMMA work.
template <bool OUT_BF16, bool TR_OUT, bool HAS_BIAS>
__global__ __launch_bounds__(256) void gemm_nt_bf16(
    const unsigned short* __restrict__ A,  long batchA, int lda,
    const unsigned short* __restrict__ Bt, long batchB, int ldb,
    void* __restrict__ Cp, long batchC, int ldc,
    const float* __restrict__ bias, float scale, int Kdim)
{
  __shared__ unsigned short As[2][128 * 32];
  __shared__ unsigned short Bs[2][128 * 32];

  const int b  = blockIdx.z;
  const int N0 = blockIdx.x * 128;
  const int M0 = blockIdx.y * 128;
  A  += (long)b * batchA;
  Bt += (long)b * batchB;

  const int tid  = threadIdx.x;
  const int lane = tid & 31;
  const int wave = tid >> 5;
  const int wm0  = (wave >> 1) * 32;   // 0,32,64,96
  const int wn0  = (wave & 1) * 64;    // 0,64
  const int fr   = lane & 15;
  const int kb   = (lane >> 4) * 8;    // K/M sub-offset for upper half-wave

  const int srow = tid >> 2;           // 0..63
  const int scol = (tid & 3) * 8;      // 0,8,16,24

  auto stage = [&](int buf, int k0) {
    for (int p = 0; p < 2; ++p) {
      const int r = srow + p * 64;
#if HAS_ASYNC_LDS
      __builtin_amdgcn_global_load_async_to_lds_b128(
          (AS1 v4i*)&A[(long)(M0 + r) * lda + k0 + scol],
          (AS3 v4i*)&As[buf][r * 32 + scol], 0, 0);
      __builtin_amdgcn_global_load_async_to_lds_b128(
          (AS1 v4i*)&Bt[(long)(N0 + r) * ldb + k0 + scol],
          (AS3 v4i*)&Bs[buf][r * 32 + scol], 0, 0);
#else
      *(uint4*)&As[buf][r * 32 + scol] =
          *(const uint4*)&A[(long)(M0 + r) * lda + k0 + scol];
      *(uint4*)&Bs[buf][r * 32 + scol] =
          *(const uint4*)&Bt[(long)(N0 + r) * ldb + k0 + scol];
#endif
    }
  };

  f32x8 acc[2][4] = {};

  int buf = 0;
  stage(0, 0);
#if HAS_ASYNC_LDS
  wait_async_zero();
#endif
  __syncthreads();

  for (int k0 = 0; k0 < Kdim; k0 += 32) {
    const int  nxt  = buf ^ 1;
    const bool more = (k0 + 32) < Kdim;
    if (more) stage(nxt, k0 + 32);

    bf16x16 af[2], bfr[4];
    for (int i = 0; i < 2; ++i)
      af[i]  = load_frag(&As[buf][(wm0 + i * 16 + fr) * 32 + kb]);
    for (int j = 0; j < 4; ++j)
      bfr[j] = load_frag(&Bs[buf][(wn0 + j * 16 + fr) * 32 + kb]);

    for (int i = 0; i < 2; ++i)
      for (int j = 0; j < 4; ++j)
        acc[i][j] = __builtin_amdgcn_wmma_f32_16x16x32_bf16(
            false, af[i], false, bfr[j], (short)0, acc[i][j], false, false);

#if HAS_ASYNC_LDS
    if (more) wait_async_zero();
#endif
    __syncthreads();
    buf = nxt;
  }

  // C/D fragment layout: VGPR r, lanes 0-15 -> M=r, N=lane;
  //                              lanes 16-31 -> M=r+8, N=lane-16.
  for (int i = 0; i < 2; ++i) {
    for (int j = 0; j < 4; ++j) {
      const int n    = N0 + wn0 + j * 16 + fr;
      const float bv = HAS_BIAS ? bias[n] : 0.f;
      const int mb   = M0 + wm0 + i * 16 + kb;
      for (int r = 0; r < 8; ++r) {
        const int m = mb + r;
        const float v = acc[i][j][r] * scale + bv;
        if constexpr (OUT_BF16) {
          unsigned short* C = (unsigned short*)Cp + (long)b * batchC;
          if constexpr (TR_OUT) C[(long)n * ldc + m] = f2bf(v);
          else                  C[(long)m * ldc + n] = f2bf(v);
        } else {
          float* C = (float*)Cp + (long)b * batchC;
          C[(long)m * ldc + n] = v;
        }
      }
    }
  }
}

// Pass 1 of softmax over the QUERY axis: per (b,k) row of scoresT compute
// rowmax and 1/rowsum (contiguous reads, tiny writes).
__global__ __launch_bounds__(256) void softmax_stats(
    const float* __restrict__ scoresT,
    float* __restrict__ rmax, float* __restrict__ rinv)
{
  const int b = blockIdx.y;
  const int k = blockIdx.x;
  const float* row = scoresT + ((long)b * SEQ + k) * SEQ;
  __shared__ float red[256];
  const int tid = threadIdx.x;

  float m = -3.402823466e+38f;
  for (int q = tid; q < SEQ; q += 256) m = fmaxf(m, row[q]);
  red[tid] = m; __syncthreads();
  for (int s = 128; s > 0; s >>= 1) {
    if (tid < s) red[tid] = fmaxf(red[tid], red[tid + s]);
    __syncthreads();
  }
  m = red[0];
  __syncthreads();

  float sum = 0.f;
  for (int q = tid; q < SEQ; q += 256) sum += __expf(row[q] - m);
  red[tid] = sum; __syncthreads();
  for (int s = 128; s > 0; s >>= 1) {
    if (tid < s) red[tid] += red[tid + s];
    __syncthreads();
  }
  if (tid == 0) {
    rmax[(long)b * SEQ + k] = m;
    rinv[(long)b * SEQ + k] = 1.f / red[0];
  }
}

// Pass 2: normalize scoresT[k][q] and write attn[q][k] (bf16) via a 32x32
// LDS tile so both the reads and the transposed writes are coalesced.
__global__ __launch_bounds__(256) void softmax_norm_transpose(
    const float* __restrict__ scoresT, const float* __restrict__ rmax,
    const float* __restrict__ rinv, unsigned short* __restrict__ attn)
{
  __shared__ float tile[32][33];
  const int b  = blockIdx.z;
  const int q0 = blockIdx.x * 32;
  const int k0 = blockIdx.y * 32;
  const int tx = threadIdx.x & 31;
  const int ty = threadIdx.x >> 5;
  const float* S = scoresT + (long)b * SEQ * SEQ;

  for (int i = ty; i < 32; i += 8) {
    const int k = k0 + i;
    const float m   = rmax[(long)b * SEQ + k];
    const float inv = rinv[(long)b * SEQ + k];
    tile[i][tx] = __expf(S[(long)k * SEQ + q0 + tx] - m) * inv;
  }
  __syncthreads();

  unsigned short* Ab = attn + (long)b * SEQ * SEQ;
  for (int i = ty; i < 32; i += 8)
    Ab[(long)(q0 + i) * SEQ + k0 + tx] = f2bf(tile[tx][i]);
}

__global__ __launch_bounds__(256) void transpose_f32_to_bf16(
    const float* __restrict__ in, unsigned short* __restrict__ out,
    int R, int Cc)   // in[R][Cc] -> out[Cc][R]
{
  __shared__ float tile[32][33];
  const int c0 = blockIdx.x * 32;
  const int r0 = blockIdx.y * 32;
  const int tx = threadIdx.x & 31;
  const int ty = threadIdx.x >> 5;
  for (int i = ty; i < 32; i += 8)
    tile[i][tx] = in[(long)(r0 + i) * Cc + c0 + tx];
  __syncthreads();
  for (int i = ty; i < 32; i += 8)
    out[(long)(c0 + i) * R + r0 + tx] = f2bf(tile[tx][i]);
}

__global__ __launch_bounds__(256) void f32_to_bf16_kernel(
    const float* __restrict__ in, unsigned short* __restrict__ out, long n)
{
  for (long i = (long)blockIdx.x * 256 + threadIdx.x; i < n;
       i += (long)gridDim.x * 256)
    out[i] = f2bf(in[i]);
}

extern "C" void kernel_launch(void* const* d_in, const int* in_sizes, int n_in,
                              void* d_out, int out_size, void* d_ws, size_t ws_size,
                              hipStream_t stream)
{
  (void)in_sizes; (void)n_in; (void)out_size; (void)ws_size;
  const float* x  = (const float*)d_in[0];
  const float* Wq = (const float*)d_in[1];
  const float* bq = (const float*)d_in[2];
  const float* Wk = (const float*)d_in[3];
  const float* bk = (const float*)d_in[4];
  const float* Wv = (const float*)d_in[5];
  const float* bv = (const float*)d_in[6];
  const float* Wo = (const float*)d_in[7];
  const float* bo = (const float*)d_in[8];
  float* out = (float*)d_out;

  char* ws = (char*)d_ws;
  size_t off = 0;
  auto take = [&](size_t bytes) -> char* {
    char* p = ws + off;
    off = (off + bytes + 255) & ~(size_t)255;
    return p;
  };

  unsigned short* xb   = (unsigned short*)take((size_t)BQ * SEQ * FDIM * 2);
  unsigned short* WqT  = (unsigned short*)take((size_t)FDIM * DK * 2);
  unsigned short* WkT  = (unsigned short*)take((size_t)FDIM * DK * 2);
  unsigned short* WvT  = (unsigned short*)take((size_t)FDIM * DK * 2);
  unsigned short* WoT  = (unsigned short*)take((size_t)DK * FDIM * 2);
  unsigned short* Qb   = (unsigned short*)take((size_t)BQ * SEQ * DK * 2);
  unsigned short* Kb   = (unsigned short*)take((size_t)BQ * SEQ * DK * 2);
  unsigned short* VT   = (unsigned short*)take((size_t)BQ * DK * SEQ * 2);
  float*          scT  = (float*)         take((size_t)BQ * SEQ * SEQ * 4);
  unsigned short* attn = (unsigned short*)take((size_t)BQ * SEQ * SEQ * 2);
  unsigned short* ctx  = (unsigned short*)take((size_t)BQ * SEQ * DK * 2);
  float*          rmax = (float*)         take((size_t)BQ * SEQ * 4);
  float*          rinv = (float*)         take((size_t)BQ * SEQ * 4);

  // 1) x -> bf16
  f32_to_bf16_kernel<<<4096, 256, 0, stream>>>(x, xb, (long)BQ * SEQ * FDIM);

  // 2) weight transposes (-> NT-form bf16 operands)
  dim3 tg(DK / 32, FDIM / 32);
  transpose_f32_to_bf16<<<tg, 256, 0, stream>>>(Wq, WqT, FDIM, DK);
  transpose_f32_to_bf16<<<tg, 256, 0, stream>>>(Wk, WkT, FDIM, DK);
  transpose_f32_to_bf16<<<tg, 256, 0, stream>>>(Wv, WvT, FDIM, DK);
  dim3 tg2(FDIM / 32, DK / 32);
  transpose_f32_to_bf16<<<tg2, 256, 0, stream>>>(Wo, WoT, DK, FDIM);

  // 3) projections: Q,K row-major bf16; V stored transposed (V^T [DK,SEQ])
  dim3 gp(DK / 128, SEQ / 128, BQ);
  gemm_nt_bf16<true, false, true><<<gp, 256, 0, stream>>>(
      xb, (long)SEQ * FDIM, FDIM, WqT, 0, FDIM,
      Qb, (long)SEQ * DK, DK, bq, 1.f, FDIM);
  gemm_nt_bf16<true, false, true><<<gp, 256, 0, stream>>>(
      xb, (long)SEQ * FDIM, FDIM, WkT, 0, FDIM,
      Kb, (long)SEQ * DK, DK, bk, 1.f, FDIM);
  gemm_nt_bf16<true, true, true><<<gp, 256, 0, stream>>>(
      xb, (long)SEQ * FDIM, FDIM, WvT, 0, FDIM,
      VT, (long)DK * SEQ, SEQ, bv, 1.f, FDIM);

  // 4) scoresT[b][k][q] = (1/sqrt(DK)) * <K[b,k], Q[b,q]>
  dim3 gs(SEQ / 128, SEQ / 128, BQ);
  gemm_nt_bf16<false, false, false><<<gs, 256, 0, stream>>>(
      Kb, (long)SEQ * DK, DK, Qb, (long)SEQ * DK, DK,
      scT, (long)SEQ * SEQ, SEQ, nullptr, 0.03125f, DK);

  // 5) softmax over query axis (rows of scoresT): stats, then coalesced
  //    normalize + transpose into attn[b][q][k] (bf16)
  softmax_stats<<<dim3(SEQ, BQ), 256, 0, stream>>>(scT, rmax, rinv);
  softmax_norm_transpose<<<dim3(SEQ / 32, SEQ / 32, BQ), 256, 0, stream>>>(
      scT, rmax, rinv, attn);

  // 6) ctx[b][q][d] = sum_k attn[q,k] * V[k,d]   (attn x V^T, NT form)
  dim3 gc(DK / 128, SEQ / 128, BQ);
  gemm_nt_bf16<true, false, false><<<gc, 256, 0, stream>>>(
      attn, (long)SEQ * SEQ, SEQ, VT, (long)DK * SEQ, SEQ,
      ctx, (long)SEQ * DK, DK, nullptr, 1.f, SEQ);

  // 7) out = ctx @ Wo + bo  (fp32 output)
  dim3 go(FDIM / 128, SEQ / 128, BQ);
  gemm_nt_bf16<false, false, true><<<go, 256, 0, stream>>>(
      ctx, (long)SEQ * DK, DK, WoT, 0, DK,
      out, (long)SEQ * FDIM, FDIM, bo, 1.f, DK);
}
